// InferCellV1_2525440770351
// MI455X (gfx1250) — compile-verified, matrix operand order checked by reading, not run on hardware
//
#include <hip/hip_runtime.h>

typedef _Float16 f16;
typedef _Float16 v16h __attribute__((ext_vector_type(16)));
typedef _Float16 v8h  __attribute__((ext_vector_type(8)));
typedef float    v8f  __attribute__((ext_vector_type(8)));

#define BN_EPS 1e-5f
#define NTOT   4194304   // 64 * 64 * 32 * 32
#define PSTRIDE 72       // padded ci-stride per (row,col) position: 16B aligned, bank-friendly

// ---------------------------------------------------------------------------
// soft channel masks: m1[c] = sum_{i >= c/8} a1[i];  cm = m1*m2
// ---------------------------------------------------------------------------
__global__ void k_prep_masks(const float* __restrict__ a1, const float* __restrict__ a2,
                             float* __restrict__ m1, float* __restrict__ cm) {
  int c = threadIdx.x;                 // 64 threads
  int g = c >> 3;
  float s1 = 0.f, s2 = 0.f;
  for (int i = g; i < 8; ++i) { s1 += a1[i]; s2 += a2[i]; }
  m1[c] = s1;
  cm[c] = s1 * s2;
}

__global__ void k_zero(float* __restrict__ p, int n) {
  int i = blockIdx.x * blockDim.x + threadIdx.x;
  if (i < n) p[i] = 0.f;
}

// ---------------------------------------------------------------------------
// Repack weights to f16, pre-swizzled into WMMA B-fragment order:
//   [chunk][nb(4)][lane(32)][h(16)]  so a lane's fragment is one contiguous 32B.
// Part-1 K order: k = tap*64 + ci           (K = 576 = 18 chunks of 32)
// Part-2 K order: k = g*96 + tap*8 + c8     (K = 768 = 24 chunks; group-major for
//                 the prefix-checkpoint trick; taps 9..11 zero-padded)
// ---------------------------------------------------------------------------
__global__ void k_prep_weights(const float* __restrict__ w10, const float* __restrict__ w11,
                               const float* __restrict__ w14, const float* __restrict__ w15,
                               const float* __restrict__ w20, const float* __restrict__ w21,
                               const float* __restrict__ w24, const float* __restrict__ w25,
                               f16* __restrict__ B1, f16* __restrict__ B2) {
  const float* W1s[4] = {w10, w11, w14, w15};
  const float* W2s[4] = {w20, w21, w24, w25};
  const int stride = gridDim.x * blockDim.x;
  for (int idx = blockIdx.x * blockDim.x + threadIdx.x; idx < 4 * 18 * 2048; idx += stride) {
    int wsel = idx / 36864, rem = idx % 36864;
    int chunk = rem >> 11, r2 = rem & 2047;
    int nb = r2 >> 9, lane = (r2 >> 4) & 31, h = r2 & 15;
    int hi = lane >= 16;
    int kk = (h < 8) ? ((hi ? 8 : 0) + h) : ((hi ? 24 : 16) + (h - 8));
    int kg = chunk * 32 + kk;
    int tap = kg >> 6, ci = kg & 63;
    int co = nb * 16 + (lane & 15);
    B1[idx] = (f16)W1s[wsel][(co * 64 + ci) * 9 + tap];
  }
  for (int idx = blockIdx.x * blockDim.x + threadIdx.x; idx < 4 * 24 * 2048; idx += stride) {
    int wsel = idx / 49152, rem = idx % 49152;
    int chunk = rem >> 11, r2 = rem & 2047;
    int nb = r2 >> 9, lane = (r2 >> 4) & 31, h = r2 & 15;
    int hi = lane >= 16;
    int kk = (h < 8) ? ((hi ? 8 : 0) + h) : ((hi ? 24 : 16) + (h - 8));
    int g = chunk / 3, sub = chunk % 3;
    int loc = sub * 32 + kk;
    int tap = loc >> 3, c8 = loc & 7;
    int ci = 8 * g + c8;
    int co = nb * 16 + (lane & 15);
    float v = (tap < 9) ? W2s[wsel][(co * 64 + ci) * 9 + tap] : 0.f;
    B2[idx] = (f16)v;
  }
}

// ---------------------------------------------------------------------------
// Part-1 conv: t = conv3x3(relu(src), W1) -> f16 + per-channel (sum,sumsq).
// Implicit GEMM, block M-tile = 256 positions (8 image rows), N = 64, K = 576.
// Wave wv owns image row wv as two 16-col M-subtiles -> 8 accumulators sharing
// 4 B fragments.  B fragments are double-buffered one chunk ahead so the
// WMMAs never wait on their own loads.
// ---------------------------------------------------------------------------
__global__ __launch_bounds__(256) void k_conv1(const float* __restrict__ src,
                                               const f16* __restrict__ Bw,
                                               f16* __restrict__ t,
                                               float* __restrict__ stats) {
  __shared__ __attribute__((aligned(32))) f16 lds_a[10 * 34 * PSTRIDE];
  __shared__ float lds_s[128];
  const int tid = threadIdx.x;
  const int n = blockIdx.x >> 2, hb = blockIdx.x & 3;

  // stage rows 8hb-1 .. 8hb+8 (fused ReLU; halo = 0); coalesced global reads
  for (int i = tid; i < 64 * 10 * 34; i += 256) {
    int ci = i / 340, rem = i % 340, r10 = rem / 34, colx = rem % 34;
    int gh = hb * 8 - 1 + r10, gw = colx - 1;
    float v = 0.f;
    if ((unsigned)gh < 32u && (unsigned)gw < 32u) {
      v = src[((n * 64 + ci) * 32 + gh) * 32 + gw];
      v = v > 0.f ? v : 0.f;
    }
    lds_a[(r10 * 34 + colx) * PSTRIDE + ci] = (f16)v;
  }
  for (int i = tid; i < 128; i += 256) lds_s[i] = 0.f;
  __syncthreads();                                   // only barrier before epilogue

  const int wv = tid >> 5, lane = tid & 31;
  const int m = lane & 15;
  const bool hi = lane >= 16;
  const int r = wv;                                  // image row (within 8-row tile)
  const int colA = m, colB = 16 + m;

  v8f acc[8] = {};                                   // [sub(2)][nb(4)]
  v16h bf0[4], bf1[4];
  #pragma unroll
  for (int nb = 0; nb < 4; ++nb)
    bf0[nb] = *(const v16h*)(Bw + nb * 512 + lane * 16);

  auto step = [&](int cc, v16h (&bu)[4], v16h (&bp)[4]) {
    // issue next chunk's B loads first (read-ahead at tail stays in ws, unused)
    #pragma unroll
    for (int nb = 0; nb < 4; ++nb)
      bp[nb] = *(const v16h*)(Bw + (size_t)(cc + 1) * 2048 + nb * 512 + lane * 16);
    int t0 = cc * 32 + (hi ? 8 : 0);
    int t1 = cc * 32 + (hi ? 24 : 16);
    int tap0 = t0 >> 6, cib0 = t0 & 63;
    int tap1 = t1 >> 6, cib1 = t1 & 63;
    int base0 = (r + tap0 / 3) * 34 + tap0 % 3;
    int base1 = (r + tap1 / 3) * 34 + tap1 % 3;
    const v8h aA0 = *(const v8h*)(lds_a + (base0 + colA) * PSTRIDE + cib0);
    const v8h aA1 = *(const v8h*)(lds_a + (base1 + colA) * PSTRIDE + cib1);
    const v8h aB0 = *(const v8h*)(lds_a + (base0 + colB) * PSTRIDE + cib0);
    const v8h aB1 = *(const v8h*)(lds_a + (base1 + colB) * PSTRIDE + cib1);
    v16h aA = __builtin_shufflevector(aA0, aA1, 0,1,2,3,4,5,6,7,8,9,10,11,12,13,14,15);
    v16h aB = __builtin_shufflevector(aB0, aB1, 0,1,2,3,4,5,6,7,8,9,10,11,12,13,14,15);
    #pragma unroll
    for (int nb = 0; nb < 4; ++nb) {
      acc[nb]     = __builtin_amdgcn_wmma_f32_16x16x32_f16(false, aA, false, bu[nb],
                                                           (short)0, acc[nb], false, false);
      acc[4 + nb] = __builtin_amdgcn_wmma_f32_16x16x32_f16(false, aB, false, bu[nb],
                                                           (short)0, acc[4 + nb], false, false);
    }
  };
  #pragma unroll 1
  for (int cc = 0; cc < 18; cc += 2) {
    step(cc, bf0, bf1);
    step(cc + 1, bf1, bf0);
  }

  // epilogue: f16 store (8 contiguous positions/lane -> b128) + LDS stat reduce
  const int gh = hb * 8 + wv;
  #pragma unroll
  for (int sub = 0; sub < 2; ++sub) {
    const int w0 = sub * 16 + (hi ? 8 : 0);
    #pragma unroll
    for (int nb = 0; nb < 4; ++nb) {
      int c = nb * 16 + m;
      v8h hv; float s = 0.f, sq = 0.f;
      #pragma unroll
      for (int rr = 0; rr < 8; ++rr) {
        float x = acc[sub * 4 + nb][rr];
        s += x; sq += x * x; hv[rr] = (f16)x;
      }
      *reinterpret_cast<v8h*>(t + (((size_t)(n * 64 + c) * 32 + gh) * 32 + w0)) = hv;
      atomicAdd(&lds_s[2 * c], s);
      atomicAdd(&lds_s[2 * c + 1], sq);
    }
  }
  __syncthreads();
  for (int i = tid; i < 128; i += 256) atomicAdd(&stats[i], lds_s[i]);
}

// per-channel BN finalize for part-1, folded with soft mask m1 into affine a,b
__global__ void k_fin1(const float* __restrict__ stats, const float* __restrict__ m1,
                       float* __restrict__ aff_a, float* __restrict__ aff_b) {
  int c = threadIdx.x;                   // 64
  const float N = 65536.f;
  float mu = stats[2 * c] / N;
  float var = stats[2 * c + 1] / N - mu * mu;
  float s = rsqrtf(var + BN_EPS);
  aff_a[c] = s * m1[c];
  aff_b[c] = -mu * s * m1[c];
}

// ---------------------------------------------------------------------------
// Part-2 conv with prefix checkpoints: group-major K; after group g the
// accumulator equals y_g = conv(h * mask_g, W2).  11 LDS rows (taps 9..11 read
// row r+3: finite values killed by zero-padded B weights).
// ---------------------------------------------------------------------------
__global__ __launch_bounds__(256) void k_conv2(const f16* __restrict__ t,
                                               const float* __restrict__ aff_a,
                                               const float* __restrict__ aff_b,
                                               const f16* __restrict__ Bw,
                                               f16* __restrict__ yck,
                                               float* __restrict__ stats2) {
  __shared__ __attribute__((aligned(32))) f16 lds_a[11 * 34 * PSTRIDE];
  __shared__ float lds_s[1024];
  const int tid = threadIdx.x;
  const int n = blockIdx.x >> 2, hb = blockIdx.x & 3;

  for (int i = tid; i < 64 * 11 * 34; i += 256) {
    int ci = i / 374, rem = i % 374, r11 = rem / 34, colx = rem % 34;
    int gh = hb * 8 - 1 + r11, gw = colx - 1;
    float v = 0.f;                                   // spatial zero-pad (post-BN)
    if ((unsigned)gh < 32u && (unsigned)gw < 32u)
      v = (float)t[((n * 64 + ci) * 32 + gh) * 32 + gw] * aff_a[ci] + aff_b[ci];
    lds_a[(r11 * 34 + colx) * PSTRIDE + ci] = (f16)v;
  }
  for (int i = tid; i < 1024; i += 256) lds_s[i] = 0.f;
  __syncthreads();

  const int wv = tid >> 5, lane = tid & 31;
  const int m = lane & 15;
  const bool hi = lane >= 16;
  const int r = wv;
  const int colA = m, colB = 16 + m;
  const int gh0 = hb * 8 + wv;

  v8f acc[8] = {};
  v16h bf0[4], bf1[4];
  #pragma unroll
  for (int nb = 0; nb < 4; ++nb)
    bf0[nb] = *(const v16h*)(Bw + nb * 512 + lane * 16);

  auto step = [&](int ch, v16h (&bu)[4], v16h (&bp)[4]) {
    const int g = ch / 3, sub = ch % 3;
    #pragma unroll
    for (int nb = 0; nb < 4; ++nb)
      bp[nb] = *(const v16h*)(Bw + (size_t)(ch + 1) * 2048 + nb * 512 + lane * 16);
    int tap0 = (sub * 32 + (hi ? 8 : 0)) >> 3;       // tap uniform per run
    int tap1 = (sub * 32 + (hi ? 24 : 16)) >> 3;
    int base0 = (r + tap0 / 3) * 34 + tap0 % 3;
    int base1 = (r + tap1 / 3) * 34 + tap1 % 3;
    const v8h aA0 = *(const v8h*)(lds_a + (base0 + colA) * PSTRIDE + 8 * g);
    const v8h aA1 = *(const v8h*)(lds_a + (base1 + colA) * PSTRIDE + 8 * g);
    const v8h aB0 = *(const v8h*)(lds_a + (base0 + colB) * PSTRIDE + 8 * g);
    const v8h aB1 = *(const v8h*)(lds_a + (base1 + colB) * PSTRIDE + 8 * g);
    v16h aA = __builtin_shufflevector(aA0, aA1, 0,1,2,3,4,5,6,7,8,9,10,11,12,13,14,15);
    v16h aB = __builtin_shufflevector(aB0, aB1, 0,1,2,3,4,5,6,7,8,9,10,11,12,13,14,15);
    #pragma unroll
    for (int nb = 0; nb < 4; ++nb) {
      acc[nb]     = __builtin_amdgcn_wmma_f32_16x16x32_f16(false, aA, false, bu[nb],
                                                           (short)0, acc[nb], false, false);
      acc[4 + nb] = __builtin_amdgcn_wmma_f32_16x16x32_f16(false, aB, false, bu[nb],
                                                           (short)0, acc[4 + nb], false, false);
    }
    if (sub == 2) {                                  // checkpoint: acc == y_g
      #pragma unroll
      for (int s2 = 0; s2 < 2; ++s2) {
        const int w0 = s2 * 16 + (hi ? 8 : 0);
        #pragma unroll
        for (int nb = 0; nb < 4; ++nb) {
          int c = nb * 16 + m;
          v8h hv; float s = 0.f, sq = 0.f;
          #pragma unroll
          for (int rr = 0; rr < 8; ++rr) {
            float x = acc[s2 * 4 + nb][rr];
            s += x; sq += x * x; hv[rr] = (f16)x;
          }
          size_t base = (((size_t)(g * 64 + n) * 64 + c) * 32 + gh0) * 32 + w0;
          *reinterpret_cast<v8h*>(yck + base) = hv;
          atomicAdd(&lds_s[g * 128 + 2 * c], s);
          atomicAdd(&lds_s[g * 128 + 2 * c + 1], sq);
        }
      }
    }
  };
  #pragma unroll 1
  for (int ch = 0; ch < 24; ch += 2) {
    step(ch, bf0, bf1);
    step(ch + 1, bf1, bf0);
  }
  __syncthreads();
  for (int i = tid; i < 1024; i += 256) atomicAdd(&stats2[i], lds_s[i]);
}

// BN finalize per (k, channel); fold alpha2: out = sum_k coef[k][c]*y_k + coefb[k][c]
__global__ void k_fin2(const float* __restrict__ stats2, const float* __restrict__ a2,
                       float* __restrict__ coef, float* __restrict__ coefb) {
  int i = threadIdx.x;                   // 512 = 8 groups x 64 channels
  int g = i >> 6;
  const float N = 65536.f;
  float mu = stats2[2 * i] / N;
  float var = stats2[2 * i + 1] / N - mu * mu;
  float s = rsqrtf(var + BN_EPS);
  coef[i]  = a2[g] * s;
  coefb[i] = -mu * a2[g] * s;
}

__global__ __launch_bounds__(256) void k_combine(const f16* __restrict__ yck,
                                                 const float* __restrict__ coef,
                                                 const float* __restrict__ coefb,
                                                 float* __restrict__ dst, int accumulate) {
  __shared__ float sc[512], scb[512];
  for (int i = threadIdx.x; i < 512; i += 256) { sc[i] = coef[i]; scb[i] = coefb[i]; }
  __syncthreads();
  int idx = blockIdx.x * 256 + threadIdx.x;      // grid covers exactly NTOT
  int c = (idx >> 10) & 63;
  float v = 0.f;
  #pragma unroll
  for (int g = 0; g < 8; ++g) {
    float y = (float)yck[(size_t)g * NTOT + idx];
    v += sc[g * 64 + c] * y + scb[g * 64 + c];
  }
  if (accumulate) dst[idx] += v;
  else            dst[idx] = v;
}

// n2 += cm[c] * avgpool3x3(n1)  (count_include_pad=False)
__global__ void k_pool_add(const float* __restrict__ n1, const float* __restrict__ cm,
                           float* __restrict__ n2) {
  int idx = blockIdx.x * 256 + threadIdx.x;
  int w = idx & 31, h = (idx >> 5) & 31;
  int base = idx - (h * 32 + w);
  float s = 0.f; int cnt = 0;
  #pragma unroll
  for (int dh = -1; dh <= 1; ++dh)
    #pragma unroll
    for (int dw = -1; dw <= 1; ++dw) {
      int hh = h + dh, ww = w + dw;
      if ((unsigned)hh < 32u && (unsigned)ww < 32u) { s += n1[base + hh * 32 + ww]; ++cnt; }
    }
  int c = (idx >> 10) & 63;
  n2[idx] += cm[c] * (s / (float)cnt);
}

__global__ void k_out_init(const float* __restrict__ x, const float* __restrict__ cm,
                           float* __restrict__ out) {
  int idx = blockIdx.x * 256 + threadIdx.x;
  int c = (idx >> 10) & 63;
  out[idx] = x[idx] * cm[c];
}

// ---------------------------------------------------------------------------
extern "C" void kernel_launch(void* const* d_in, const int* in_sizes, int n_in,
                              void* d_out, int out_size, void* d_ws, size_t ws_size,
                              hipStream_t stream) {
  (void)in_sizes; (void)n_in; (void)out_size; (void)ws_size;
  const float* x  = (const float*)d_in[0];
  const float* a1 = (const float*)d_in[1];
  const float* a2 = (const float*)d_in[2];
  const float* W1[4] = {(const float*)d_in[3], (const float*)d_in[5],
                        (const float*)d_in[7], (const float*)d_in[9]};
  const float* W2[4] = {(const float*)d_in[4], (const float*)d_in[6],
                        (const float*)d_in[8], (const float*)d_in[10]};
  float* out = (float*)d_out;

  // workspace carve-up
  char*  wsb   = (char*)d_ws;
  float* sm    = (float*)wsb;
  float* m1    = sm;          float* cm    = sm + 64;
  float* aff_a = sm + 128;    float* aff_b = sm + 192;
  float* stats1= sm + 256;    float* stats2= sm + 384;   // 128 + 1024 floats
  float* coef  = sm + 1408;   float* coefb = sm + 1920;
  f16*   B1    = (f16*)(wsb + 16384);                    // 4 x 18 x 2048 f16 (swizzled)
  f16*   B2    = (f16*)(wsb + 311296);                   // 4 x 24 x 2048 f16 (swizzled)
  f16*   tbuf  = (f16*)(wsb + 704512);                   // [64,64,32,32] f16
  f16*   yck   = (f16*)(wsb + 9093120);                  // [8,64,64,32,32] f16
  float* n1    = (float*)(wsb + 76201984);
  float* n2    = (float*)(wsb + 92979200);

  k_prep_masks<<<1, 64, 0, stream>>>(a1, a2, m1, cm);
  k_prep_weights<<<512, 256, 0, stream>>>(W1[0], W1[1], W1[2], W1[3],
                                          W2[0], W2[1], W2[2], W2[3], B1, B2);

  auto run_nc = [&](const float* src, int widx, float* dst, int accumulate) {
    k_zero<<<5, 256, 0, stream>>>(stats1, 1152);          // stats1 + stats2 (contiguous)
    k_conv1<<<256, 256, 0, stream>>>(src, B1 + (size_t)widx * 36864, tbuf, stats1);
    k_fin1<<<1, 64, 0, stream>>>(stats1, m1, aff_a, aff_b);
    k_conv2<<<256, 256, 0, stream>>>(tbuf, aff_a, aff_b,
                                     B2 + (size_t)widx * 49152, yck, stats2);
    k_fin2<<<1, 512, 0, stream>>>(stats2, a2, coef, coefb);
    k_combine<<<NTOT / 256, 256, 0, stream>>>(yck, coef, coefb, dst, accumulate);
  };

  run_nc(x,  0, n1, 0);                 // n1 = nor_conv(x; W1_0, W2_0)
  run_nc(x,  1, n2, 0);                 // n2 = nor_conv(x; W1_1, W2_1)
  k_pool_add<<<NTOT / 256, 256, 0, stream>>>(n1, cm, n2);  // + avgpool(n1)*cm
  k_out_init<<<NTOT / 256, 256, 0, stream>>>(x, cm, out);  // n3 = x*cm
  run_nc(n1, 2, out, 1);                // + nor_conv(n1; W1_4, W2_4)
  run_nc(n2, 3, out, 1);                // + nor_conv(n2; W1_5, W2_5)
}
